// Attn_Decoder_38062000177828
// MI455X (gfx1250) — matrix operand order repaired and checked by reading, not compiled, not run on hardware
//
#include <hip/hip_runtime.h>

// ---------------------------------------------------------------------------
// CDNA5 (gfx1250) implementation of the attention-decoder block.
// bf16 NHWC activations + [N][K] bf16 weights, every conv as an implicit GEMM
// on v_wmma_f32_16x16x32_bf16 (f32 accumulate). GEMM kernels are templated on
// K and N-tiles-per-wave so accumulators are statically indexed (no v_movrel).
// ---------------------------------------------------------------------------

typedef __attribute__((ext_vector_type(16))) __bf16 v16bf;
typedef __attribute__((ext_vector_type(8)))  __bf16 v8bf;
typedef __attribute__((ext_vector_type(8)))  float  v8f;

#define HSZ    192
#define HHPIX  (HSZ*HSZ)          // 36864
#define NB     4
#define PTOT   (NB*HHPIX)         // 147456 pixels
#define GD     24
#define NGR    (GD*GD)            // 576
#define SPDD   384

// ---------------------------- prep kernels ---------------------------------

// NCHW f32 -> NHWC bf16 (Xt [P][96]); also fill channels 96..191 of cat [P][192]
__global__ void k_prep_fea(const float* __restrict__ X,
                           __bf16* __restrict__ Xt,
                           __bf16* __restrict__ cat192) {
  int i = blockIdx.x * 256 + threadIdx.x;            // over NB*96*HHPIX (exact)
  int b = i / (96 * HHPIX);
  int r = i % (96 * HHPIX);
  int c = r / HHPIX;
  int pix = r % HHPIX;
  float v = X[i];
  size_t p = (size_t)b * HHPIX + pix;
  __bf16 bv = (__bf16)v;
  Xt[p * 96 + c] = bv;
  cat192[p * 192 + 96 + c] = bv;
}

// [N][K] f32 -> [Npad][K] bf16 (zero pad rows n>=N)
__global__ void k_prep_w1x1(const float* __restrict__ W, __bf16* __restrict__ Wo,
                            int N, int K, int total) {
  int i = blockIdx.x * 256 + threadIdx.x;
  if (i >= total) return;
  int n = i / K;
  Wo[i] = (n < N) ? (__bf16)W[i] : (__bf16)0.0f;
}

// [96][96][3][3] f32 -> [96][864] bf16 with k = tap*96 + c  (tap = dy*3+dx)
__global__ void k_prep_w3x3(const float* __restrict__ W, __bf16* __restrict__ Wo) {
  int i = blockIdx.x * 256 + threadIdx.x;
  if (i >= 96 * 864) return;
  int n = i / 864, k = i % 864;
  int tap = k / 96, c = k % 96;
  Wo[i] = (__bf16)W[((size_t)n * 96 + c) * 9 + tap];
}

// ---------------------- LayerNorm(384) + Linear(384->96) -------------------
// one block (128 thr) per (b, n) row; out fea_v [B][96][576] f32
__global__ void k_ln_linear(const float* __restrict__ Xin,   // [B][576][384]
                            const float* __restrict__ g, const float* __restrict__ beta,
                            const float* __restrict__ W,     // [96][384]
                            const float* __restrict__ bw,    // [96]
                            float* __restrict__ out) {
  __shared__ float xs[SPDD];
  __shared__ float red[128];
  __shared__ float stats[2];
  int row = blockIdx.x;                 // b*576 + n
  int tid = threadIdx.x;
  const float* xr = Xin + (size_t)row * SPDD;
  float s = 0.f;
  for (int i = tid; i < SPDD; i += 128) { float v = xr[i]; xs[i] = v; s += v; }
  red[tid] = s; __syncthreads();
  for (int st = 64; st > 0; st >>= 1) { if (tid < st) red[tid] += red[tid + st]; __syncthreads(); }
  if (tid == 0) stats[0] = red[0] * (1.0f / SPDD);
  __syncthreads();
  float mu = stats[0];
  float sq = 0.f;
  for (int i = tid; i < SPDD; i += 128) { float d = xs[i] - mu; sq += d * d; }
  red[tid] = sq; __syncthreads();
  for (int st = 64; st > 0; st >>= 1) { if (tid < st) red[tid] += red[tid + st]; __syncthreads(); }
  if (tid == 0) stats[1] = rsqrtf(red[0] * (1.0f / SPDD) + 1e-5f);
  __syncthreads();
  float rv = stats[1];
  for (int i = tid; i < SPDD; i += 128) xs[i] = (xs[i] - mu) * rv * g[i] + beta[i];
  __syncthreads();
  if (tid < 96) {
    const float* wr = W + (size_t)tid * SPDD;
    float a = bw[tid];
    for (int i = 0; i < SPDD; ++i) a += xs[i] * wr[i];
    int b = row / NGR, n = row % NGR;
    out[((size_t)b * 96 + tid) * NGR + n] = a;
  }
}

// ------------------- local window attention (9-tap einsum) -----------------
// out[p][ch] = BN( sum_c Q[p][c*12+h] * V[b][h*8+d][clamp(gy+dy)][clamp(gx+dx)] )
// ch = d*12 + h ; written as bf16 (input of the proj_f GEMM).
__global__ void k_attn(const float* __restrict__ Q,     // [P][108]
                       const float* __restrict__ V,     // [B][96][24][24]
                       const float* __restrict__ bns, const float* __restrict__ bnb,
                       __bf16* __restrict__ out) {      // [P][96]
  int t = blockIdx.x * 256 + threadIdx.x;
  if (t >= PTOT * 96) return;
  int p = t / 96, ch = t % 96;
  int d = ch / 12, h = ch % 12;
  int b = p / HHPIX; int rem = p % HHPIX;
  int y = rem / HSZ, x = rem % HSZ;
  int gy = y >> 3, gx = x >> 3;
  const float* vbase = V + ((size_t)b * 96 + (h * 8 + d)) * NGR;
  const float* qbase = Q + (size_t)p * 108 + h;
  float acc = 0.f;
#pragma unroll
  for (int c = 0; c < 9; ++c) {
    int dy = c / 3 - 1, dx = c % 3 - 1;
    int yy = gy + dy; yy = yy < 0 ? 0 : (yy > GD - 1 ? GD - 1 : yy);
    int xx = gx + dx; xx = xx < 0 ? 0 : (xx > GD - 1 ? GD - 1 : xx);
    acc += qbase[c * 12] * vbase[yy * GD + xx];
  }
  out[(size_t)p * 96 + ch] = (__bf16)(acc * bns[ch] + bnb[ch]);
}

// --------------------------- 1x1 conv as WMMA GEMM -------------------------
// M tile = 64 pixels/block, 8 waves: wave = (wm 0..3) x (wn 0..1).
// Compile-time K and NT (N-tiles per wave-group) => statically indexed accs.
// Covers n in [0, 2*NT*16); W must be zero-padded to >= 2*NT*16 rows.
template <int K, int NT>
__global__ void k_gemm1x1_wmma(const __bf16* __restrict__ X, int ldx,
                               const __bf16* __restrict__ W,   // [>=2*NT*16][K]
                               const float* __restrict__ bias,
                               const float* __restrict__ bns, const float* __restrict__ bnb,
                               int relu, int N,
                               float* __restrict__ of32, int ldo,
                               __bf16* __restrict__ obf, int ldob, int choff) {
  int lane = threadIdx.x & 31;
  int wid  = threadIdx.x >> 5;
  int wm = wid & 3, wn = wid >> 2;
  int nfirst = wn * NT;
  size_t p0 = (size_t)blockIdx.x * 64 + (size_t)wm * 16;
  int ml = lane & 15;
  int hi = lane >> 4;
  const __bf16* arow = X + (p0 + ml) * (size_t)ldx + hi * 8;

  v8f zero = {};
  v8f acc[NT];
#pragma unroll
  for (int t = 0; t < NT; ++t) acc[t] = zero;

#pragma unroll
  for (int kb = 0; kb < K; kb += 32) {
    union { v16bf v; v8bf h[2]; } ua;
    ua.h[0] = *(const v8bf*)(arow + kb);
    ua.h[1] = *(const v8bf*)(arow + kb + 16);
    v16bf a = ua.v;
#pragma unroll
    for (int t = 0; t < NT; ++t) {
      int nb = (nfirst + t) * 16;
      const __bf16* brow = W + (size_t)(nb + ml) * K + kb + hi * 8;
      union { v16bf v; v8bf h[2]; } ub;
      ub.h[0] = *(const v8bf*)(brow);
      ub.h[1] = *(const v8bf*)(brow + 16);
      acc[t] = __builtin_amdgcn_wmma_f32_16x16x32_bf16(
          false, a, false, ub.v, (short)0, acc[t], false, false);
    }
  }

#pragma unroll
  for (int t = 0; t < NT; ++t) {
    int n = (nfirst + t) * 16 + ml;
    if (n >= N) continue;
    float bsum = bias ? bias[n] : 0.f;
    float sc = bns ? bns[n] : 1.f;
    float sh = bnb ? bnb[n] : 0.f;
#pragma unroll
    for (int i = 0; i < 8; ++i) {
      size_t prow = p0 + i + hi * 8;
      float v = (acc[t][i] + bsum) * sc + sh;
      if (relu) v = fmaxf(v, 0.f);
      if (of32) of32[prow * (size_t)ldo + n] = v;
      if (obf)  obf[prow * (size_t)ldob + choff + n] = (__bf16)v;
    }
  }
}

// --------------------------- 3x3 conv as WMMA GEMM -------------------------
// K = 9 taps * 96 ch, zero padding. N=96 fixed. Input NHWC bf16 [P][96].
// Outputs: optional NCHW f32 (final outputs) and/or NHWC bf16 (next stage),
// bf16 side optionally accumulates (used to build fea_patch+fea_sp).
__global__ void k_conv3x3_wmma(const __bf16* __restrict__ X,   // [P][96]
                               const __bf16* __restrict__ W,   // [96][864]
                               const float* __restrict__ bias,
                               const float* __restrict__ bns, const float* __restrict__ bnb,
                               int relu,
                               float* __restrict__ of32_nchw,  // [B][96][H][H] or null
                               __bf16* __restrict__ obf,       // [P][96] or null
                               int accum_bf) {
  int lane = threadIdx.x & 31;
  int wid  = threadIdx.x >> 5;
  int wm = wid & 3, wn = wid >> 2;
  int nfirst = wn * 3;                       // 3 N-tiles per wave (N=96)
  size_t p0 = (size_t)blockIdx.x * 64 + (size_t)wm * 16;
  int ml = lane & 15;
  int hi = lane >> 4;
  size_t p = p0 + ml;
  int b = (int)(p / HHPIX); int rem = (int)(p % HHPIX);
  int y = rem / HSZ, x = rem % HSZ;

  v8f zero = {};
  v8f acc[3]; acc[0] = zero; acc[1] = zero; acc[2] = zero;

#pragma unroll
  for (int tap = 0; tap < 9; ++tap) {
    int dy = tap / 3 - 1, dx = tap % 3 - 1;
    int yy = y + dy, xx = x + dx;
    bool inb = (yy >= 0) && (yy < HSZ) && (xx >= 0) && (xx < HSZ);
    const __bf16* arow =
        X + (((size_t)b * HSZ + (inb ? yy : 0)) * HSZ + (inb ? xx : 0)) * 96 + hi * 8;
    if (tap < 8) {  // prefetch next tap row into L2/L0
      int dy2 = (tap + 1) / 3 - 1, dx2 = (tap + 1) % 3 - 1;
      int y2 = y + dy2, x2 = x + dx2;
      if (y2 >= 0 && y2 < HSZ && x2 >= 0 && x2 < HSZ)
        __builtin_prefetch(X + (((size_t)b * HSZ + y2) * HSZ + x2) * 96, 0, 3);
    }
#pragma unroll
    for (int cb = 0; cb < 96; cb += 32) {
      v16bf a;
      if (inb) {
        union { v16bf v; v8bf h[2]; } ua;
        ua.h[0] = *(const v8bf*)(arow + cb);
        ua.h[1] = *(const v8bf*)(arow + cb + 16);
        a = ua.v;
      } else {
        union { v16bf v; __bf16 e[16]; } uz;
#pragma unroll
        for (int j = 0; j < 16; ++j) uz.e[j] = (__bf16)0.0f;
        a = uz.v;
      }
      int kb = tap * 96 + cb;
#pragma unroll
      for (int t = 0; t < 3; ++t) {
        int nb = (nfirst + t) * 16;
        const __bf16* brow = W + (size_t)(nb + ml) * 864 + kb + hi * 8;
        union { v16bf v; v8bf h[2]; } ub;
        ub.h[0] = *(const v8bf*)(brow);
        ub.h[1] = *(const v8bf*)(brow + 16);
        acc[t] = __builtin_amdgcn_wmma_f32_16x16x32_bf16(
            false, a, false, ub.v, (short)0, acc[t], false, false);
      }
    }
  }

  // store: C layout n = lane%16, m = i + hi*8  (16-pixel block stays in one row)
  int b0 = (int)(p0 / HHPIX); int r0 = (int)(p0 % HHPIX);
  int y0 = r0 / HSZ, x0 = r0 % HSZ;
#pragma unroll
  for (int t = 0; t < 3; ++t) {
    int n = (nfirst + t) * 16 + ml;
    float bsum = bias[n];
    float sc = bns[n], sh = bnb[n];
#pragma unroll
    for (int i = 0; i < 8; ++i) {
      int xo = x0 + i + hi * 8;
      float v = (acc[t][i] + bsum) * sc + sh;
      if (relu) v = fmaxf(v, 0.f);
      if (of32_nchw)
        of32_nchw[(((size_t)b0 * 96 + n) * HHPIX) + (size_t)y0 * HSZ + xo] = v;
      if (obf) {
        size_t prow = p0 + i + hi * 8;
        float vb = v;
        if (accum_bf) vb += (float)obf[prow * 96 + n];
        obf[prow * 96 + n] = (__bf16)vb;
      }
    }
  }
}

// ---------------------------------------------------------------------------

extern "C" void kernel_launch(void* const* d_in, const int* in_sizes, int n_in,
                              void* d_out, int out_size, void* d_ws, size_t ws_size,
                              hipStream_t stream) {
  // setup_inputs() dict (insertion) order:
  // 0 temp_fea, 1 temp_sp, 2 temp_patch, then params in insertion order:
  const float* temp_fea   = (const float*)d_in[0];
  const float* temp_sp    = (const float*)d_in[1];
  const float* temp_patch = (const float*)d_in[2];
  const float* w_t1  = (const float*)d_in[3];   const float* b_t1  = (const float*)d_in[4];
  const float* w_t2  = (const float*)d_in[5];   const float* b_t2  = (const float*)d_in[6];
  const float* ln_g_p = (const float*)d_in[7];  const float* ln_b_p = (const float*)d_in[8];
  const float* w_pp  = (const float*)d_in[9];   const float* b_pp  = (const float*)d_in[10];
  const float* ln_g_s = (const float*)d_in[11]; const float* ln_b_s = (const float*)d_in[12];
  const float* w_ps  = (const float*)d_in[13];  const float* b_ps  = (const float*)d_in[14];
  const float* w_fp  = (const float*)d_in[15];  const float* b_fp  = (const float*)d_in[16];
  const float* w_fs  = (const float*)d_in[17];  const float* b_fs  = (const float*)d_in[18];
  const float* w_mfp1 = (const float*)d_in[19]; const float* b_mfp1 = (const float*)d_in[20];
  const float* w_mfp2 = (const float*)d_in[21]; const float* b_mfp2 = (const float*)d_in[22];
  const float* w_mfs1 = (const float*)d_in[23]; const float* b_mfs1 = (const float*)d_in[24];
  const float* w_mfs2 = (const float*)d_in[25]; const float* b_mfs2 = (const float*)d_in[26];
  const float* w_ffn1 = (const float*)d_in[27]; const float* b_ffn1 = (const float*)d_in[28];
  const float* w_ffn2 = (const float*)d_in[29]; const float* b_ffn2 = (const float*)d_in[30];
  const float* bn_s = (const float*)d_in[31];   const float* bn_b = (const float*)d_in[32];

  float* out = (float*)d_out;                       // y | fea_patch | fea_sp
  const size_t REG = (size_t)PTOT * 96;
  float* out_y  = out;
  float* out_br[2] = { out + REG, out + 2 * REG };  // fea_patch, fea_sp

  // workspace carve-up
  char* ws = (char*)d_ws; size_t off = 0;
  auto carve = [&](size_t bytes) { void* q = ws + off; off += (bytes + 255) & ~(size_t)255; return q; };
  __bf16* Xt      = (__bf16*)carve((size_t)PTOT * 96 * 2);    // temp_fea NHWC bf16
  __bf16* cat     = (__bf16*)carve((size_t)PTOT * 192 * 2);   // [proj_f out | temp_fea]
  __bf16* attnb   = (__bf16*)carve((size_t)PTOT * 96 * 2);    // BN'd attention out
  float*  Q       = (float*) carve((size_t)PTOT * 108 * 4);   // proj_q out
  __bf16* m1o     = (__bf16*)carve((size_t)PTOT * 96 * 2);
  __bf16* sumb    = (__bf16*)carve((size_t)PTOT * 96 * 2);    // fea_patch+fea_sp
  __bf16* f1o     = (__bf16*)carve((size_t)PTOT * 96 * 2);
  float*  feaV    = (float*) carve((size_t)NB * 96 * NGR * 4);
  __bf16* wq      = (__bf16*)carve((size_t)128 * 96 * 2);     // padded to 128 rows
  __bf16* wf      = (__bf16*)carve((size_t)96 * 96 * 2);
  __bf16* wm1     = (__bf16*)carve((size_t)96 * 192 * 2);
  __bf16* wc3     = (__bf16*)carve((size_t)96 * 864 * 2);
  (void)n_in; (void)in_sizes; (void)out_size; (void)ws_size;

  const int GEMM_BLOCKS = PTOT / 64;                // 2304
  dim3 blk256(256);

  // one-time input transpose
  k_prep_fea<<<dim3((NB * 96 * HHPIX) / 256), blk256, 0, stream>>>(temp_fea, Xt, cat);

  // per-branch parameter tables (0 = patch, 1 = sp)
  const float* kv[2]   = { temp_patch, temp_sp };
  const float* lnG[2]  = { ln_g_p, ln_g_s };
  const float* lnB[2]  = { ln_b_p, ln_b_s };
  const float* lw[2]   = { w_pp, w_ps };
  const float* lbias[2]= { b_pp, b_ps };
  const float* qw[2]   = { w_t1, w_t2 };
  const float* qb[2]   = { b_t1, b_t2 };
  const float* fw[2]   = { w_fp, w_fs };
  const float* fb[2]   = { b_fp, b_fs };
  const float* m1w[2]  = { w_mfp1, w_mfs1 };
  const float* m1b[2]  = { b_mfp1, b_mfs1 };
  const float* m2w[2]  = { w_mfp2, w_mfs2 };
  const float* m2b[2]  = { b_mfp2, b_mfs2 };
  const int bnf_row[2] = { 0, 1 };
  const int bn1_row[2] = { 2, 4 };
  const int bn2_row[2] = { 3, 5 };

  for (int br = 0; br < 2; ++br) {
    // LayerNorm + Linear -> fea_v [B][96][24][24]
    k_ln_linear<<<dim3(NB * NGR), dim3(128), 0, stream>>>(
        kv[br], lnG[br], lnB[br], lw[br], lbias[br], feaV);

    // proj_q : conv1x1 96 -> 108  (NT=4 covers n<128; W zero-padded)
    k_prep_w1x1<<<dim3((128 * 96 + 255) / 256), blk256, 0, stream>>>(qw[br], wq, 108, 96, 128 * 96);
    k_gemm1x1_wmma<96, 4><<<dim3(GEMM_BLOCKS), blk256, 0, stream>>>(
        Xt, 96, wq, qb[br], nullptr, nullptr, 0, 108, Q, 108, nullptr, 0, 0);

    // local window attention (+ folded bnf affine) -> bf16
    k_attn<<<dim3((PTOT * 96) / 256), blk256, 0, stream>>>(
        Q, feaV, bn_s + bnf_row[br] * 96, bn_b + bnf_row[br] * 96, attnb);

    // proj_f : conv1x1 96 -> 96, write into cat channels [0,96)
    k_prep_w1x1<<<dim3((96 * 96 + 255) / 256), blk256, 0, stream>>>(fw[br], wf, 96, 96, 96 * 96);
    k_gemm1x1_wmma<96, 3><<<dim3(GEMM_BLOCKS), blk256, 0, stream>>>(
        attnb, 96, wf, fb[br], nullptr, nullptr, 0, 96,
        nullptr, 0, cat, 192, 0);

    // m1 : conv1x1 192 -> 96 + BN + ReLU
    k_prep_w1x1<<<dim3((96 * 192 + 255) / 256), blk256, 0, stream>>>(m1w[br], wm1, 96, 192, 96 * 192);
    k_gemm1x1_wmma<192, 3><<<dim3(GEMM_BLOCKS), blk256, 0, stream>>>(
        cat, 192, wm1, m1b[br], bn_s + bn1_row[br] * 96, bn_b + bn1_row[br] * 96, 1,
        96, nullptr, 0, m1o, 96, 0);

    // m2 : conv3x3 + BN  -> fea_patch/fea_sp (NCHW f32) and running sum (bf16)
    k_prep_w3x3<<<dim3((96 * 864 + 255) / 256), blk256, 0, stream>>>(m2w[br], wc3);
    k_conv3x3_wmma<<<dim3(GEMM_BLOCKS), blk256, 0, stream>>>(
        m1o, wc3, m2b[br], bn_s + bn2_row[br] * 96, bn_b + bn2_row[br] * 96, 0,
        out_br[br], sumb, br);
  }

  // FFN: two conv3x3 + BN + ReLU on (fea_patch + fea_sp)
  k_prep_w3x3<<<dim3((96 * 864 + 255) / 256), blk256, 0, stream>>>(w_ffn1, wc3);
  k_conv3x3_wmma<<<dim3(GEMM_BLOCKS), blk256, 0, stream>>>(
      sumb, wc3, b_ffn1, bn_s + 6 * 96, bn_b + 6 * 96, 1, nullptr, f1o, 0);

  k_prep_w3x3<<<dim3((96 * 864 + 255) / 256), blk256, 0, stream>>>(w_ffn2, wc3);
  k_conv3x3_wmma<<<dim3(GEMM_BLOCKS), blk256, 0, stream>>>(
      f1o, wc3, b_ffn2, bn_s + 7 * 96, bn_b + 7 * 96, 1, out_y, nullptr, 0);
}
